// CustomNetwork_37340445671627
// MI455X (gfx1250) — matrix operand order repaired and checked by reading, not compiled, not run on hardware
//
#include <hip/hip_runtime.h>
#include <hip/hip_bf16.h>

typedef __attribute__((ext_vector_type(16))) __bf16 v16bf;
typedef __attribute__((ext_vector_type(8)))  float  v8f;

#define Bq 64
#define NNq 1000
#define NVq 10
#define Hq 128
#define Tq 8
#define Oq 64
#define ENq 512000
#define EVq 2560

// ---------------- conversion / packing ----------------
__global__ void k_convert_pad(const float* __restrict__ src, int srcK,
                              __bf16* __restrict__ dst, int Kp, int total) {
  int i = blockIdx.x * blockDim.x + threadIdx.x;
  if (i >= total) return;
  int k = i % Kp;
  int r = i / Kp;
  float v = (k < srcK) ? src[(size_t)r * srcK + k] : 0.f;
  dst[i] = (__bf16)v;
}

// W stored [K,N] row-major -> WT[n*Kp+k] = W[k*N+n], zero-pad k>=K
__global__ void k_pack_wt(const float* __restrict__ W, int K, int N,
                          __bf16* __restrict__ WT, int Kp) {
  int i = blockIdx.x * blockDim.x + threadIdx.x;
  if (i >= N * Kp) return;
  int n = i / Kp, k = i % Kp;
  WT[i] = (__bf16)(k < K ? W[(size_t)k * N + n] : 0.f);
}

// src row-major rows x stride, take K cols at offset -> WT[r*K+k]
__global__ void k_pack_rows(const float* __restrict__ src, int stride, int off,
                            int rows, int K, __bf16* __restrict__ dst) {
  int i = blockIdx.x * blockDim.x + threadIdx.x;
  if (i >= rows * K) return;
  int r = i / K, k = i % K;
  dst[i] = (__bf16)src[(size_t)r * stride + off + k];
}

__global__ void k_fill(float* p, float v, int n) {
  int i = blockIdx.x * blockDim.x + threadIdx.x;
  if (i < n) p[i] = v;
}
__global__ void k_init_bias(float* __restrict__ out, const float* __restrict__ b, int total) {
  int i = blockIdx.x * blockDim.x + threadIdx.x;
  if (i < total) out[i] = b[i & (Hq - 1)];
}

// ---------------- WMMA GEMM: C[M,128] = A[M,Kp](bf16) x B, B given as WT[n][k] ----------------
// block = 256 threads (8 waves), block covers 128 rows, wave covers 16 rows x 128 cols.
__global__ void k_wmma_gemm(const __bf16* __restrict__ A, const __bf16* __restrict__ WT,
                            float* __restrict__ C, int M, int Kp) {
  __shared__ __align__(16) __bf16 sWT[128 * 128];
  const int tid  = threadIdx.x;
  const int wave = tid >> 5;
  const int lane = tid & 31;
  const int half = lane >> 4;     // 0 = lanes 0-15, 1 = lanes 16-31
  const int l16  = lane & 15;
  const int m0   = blockIdx.x * 128 + wave * 16;

  // cooperative load of WT (128 x Kp bf16) into LDS
  {
    const int nvec = (128 * Kp) >> 3;   // uint4 count
    const uint4* s = (const uint4*)WT;
    uint4* d = (uint4*)sWT;
    for (int i = tid; i < nvec; i += 256) d[i] = s[i];
  }
  __syncthreads();

  const __bf16* arow = A + (size_t)(m0 + l16) * Kp;

  v8f acc[8] = {};
  for (int kt = 0; kt < Kp; kt += 32) {
    union { v16bf v; uint4 u[2]; } af;
    // 16-bit A layout: lanes 0-15 hold K {0..7,16..23}; lanes 16-31 hold K {8..15,24..31}
    const int c0 = kt + (half ? 8 : 0);
    af.u[0] = *(const uint4*)(arow + c0);
    af.u[1] = *(const uint4*)(arow + c0 + 16);
#pragma unroll
    for (int nt = 0; nt < 8; ++nt) {
      union { v16bf v; uint4 u[2]; } bfr;
      // 16-bit B layout: lanes 0-15 hold K {0..7,16..23}; lanes 16-31 hold K {8..15,24..31} of this k-tile
      const __bf16* brow = sWT + (size_t)(nt * 16 + l16) * Kp + kt + (half ? 8 : 0);
      bfr.u[0] = *(const uint4*)(brow);
      bfr.u[1] = *(const uint4*)(brow + 16);
      acc[nt] = __builtin_amdgcn_wmma_f32_16x16x32_bf16(
          false, af.v, false, bfr.v, (short)0, acc[nt], false, false);
    }
  }
  // C/D layout: VGPR r -> row m0 + r + half*8, col n0 + l16
#pragma unroll
  for (int nt = 0; nt < 8; ++nt) {
    const int col = nt * 16 + l16;
#pragma unroll
    for (int r = 0; r < 8; ++r) {
      const int row = m0 + (half ? 8 + r : r);
      C[(size_t)row * 128 + col] = acc[nt][r];
    }
  }
}

// ---------------- GAT pieces ----------------
__global__ void k_node_dots(const float* __restrict__ z, const float* __restrict__ al,
                            const float* __restrict__ ar, float* __restrict__ el,
                            float* __restrict__ er, int M) {
  int wid = (blockIdx.x * blockDim.x + threadIdx.x) >> 5;
  int lane = threadIdx.x & 31;
  if (wid >= M) return;
  const float* row = z + (size_t)wid * Hq;
  float a = 0.f, b = 0.f;
  for (int j = lane; j < Hq; j += 32) { float v = row[j]; a += v * al[j]; b += v * ar[j]; }
  for (int o = 16; o > 0; o >>= 1) { a += __shfl_down(a, o, 32); b += __shfl_down(b, o, 32); }
  if (lane == 0) { el[wid] = a; er[wid] = b; }
}

__device__ inline void atomicMaxF(float* addr, float val) {
  int cur = __float_as_int(*addr);
  while (val > __int_as_float(cur)) {
    int prev = atomicCAS((int*)addr, cur, __float_as_int(val));
    if (prev == cur) break;
    cur = prev;
  }
}

__global__ void k_edge_scores(const int* __restrict__ src, const int* __restrict__ dst,
                              const float* __restrict__ el, const float* __restrict__ er,
                              float* __restrict__ s, float* __restrict__ emax, int E) {
  int e = blockIdx.x * blockDim.x + threadIdx.x;
  if (e >= E) return;
  float x = el[src[e]] + er[dst[e]];
  x = (x >= 0.f) ? x : 0.2f * x;      // leaky_relu 0.2
  s[e] = x;
  atomicMaxF(&emax[dst[e]], x);
}

__global__ void k_edge_exp(const int* __restrict__ dst, float* __restrict__ s,
                           const float* __restrict__ emax, float* __restrict__ denom, int E) {
  int e = blockIdx.x * blockDim.x + threadIdx.x;
  if (e >= E) return;
  int d = dst[e];
  float ee = __expf(s[e] - emax[d]);
  s[e] = ee;
  atomicAdd(&denom[d], ee);
}

__global__ void k_edge_aggregate(const int* __restrict__ src, const int* __restrict__ dst,
                                 const float* __restrict__ s, const float* __restrict__ denom,
                                 const float* __restrict__ z, float* __restrict__ out, int E) {
  int wid = (blockIdx.x * blockDim.x + threadIdx.x) >> 5;
  int lane = threadIdx.x & 31;
  if (wid >= E) return;
  int sn = src[wid], dn = dst[wid];
  float w = s[wid] / denom[dn];
  const float* zr = z + (size_t)sn * Hq;
  float* orow = out + (size_t)dn * Hq;
  for (int j = lane; j < Hq; j += 32) atomicAdd(&orow[j], zr[j] * w);
}

// ---------------- decoder ----------------
__global__ void k_h_init(float* __restrict__ h, const float* __restrict__ h0) {
  int i = blockIdx.x * blockDim.x + threadIdx.x;
  if (i < Bq * Hq) h[i] = h0[i & (Hq - 1)];
}

__global__ void k_gru_gates(const float* __restrict__ hv3, int t, const float* __restrict__ h,
                            const float* __restrict__ W_ih, const float* __restrict__ W_hh,
                            const float* __restrict__ b_ih, const float* __restrict__ b_hh,
                            float* __restrict__ gi, float* __restrict__ gh) {
  int id = blockIdx.x * blockDim.x + threadIdx.x;
  if (id >= Bq * 3 * Hq) return;
  int b = id / (3 * Hq), j = id % (3 * Hq);
  const float* x  = hv3 + (size_t)(b * NVq + t) * Hq;
  const float* hr = h + (size_t)b * Hq;
  const float* wi = W_ih + (size_t)j * Hq;
  const float* wh = W_hh + (size_t)j * Hq;
  float ai = b_ih[j], ah = b_hh[j];
  for (int k = 0; k < Hq; ++k) { ai += x[k] * wi[k]; ah += hr[k] * wh[k]; }
  gi[id] = ai; gh[id] = ah;
}

__global__ void k_gru_combine(const float* __restrict__ gi, const float* __restrict__ gh,
                              float* __restrict__ h) {
  int id = blockIdx.x * blockDim.x + threadIdx.x;
  if (id >= Bq * Hq) return;
  int b = id / Hq, j = id % Hq;
  const float* gib = gi + (size_t)b * 3 * Hq;
  const float* ghb = gh + (size_t)b * 3 * Hq;
  float r  = 1.f / (1.f + __expf(-(gib[j] + ghb[j])));
  float zz = 1.f / (1.f + __expf(-(gib[Hq + j] + ghb[Hq + j])));
  float n  = tanhf(gib[2 * Hq + j] + r * ghb[2 * Hq + j]);
  h[id] = (1.f - zz) * n + zz * h[id];
}

__global__ void k_q_proj(const float* __restrict__ W_attn, const float* __restrict__ h,
                         float* __restrict__ q) {
  int id = blockIdx.x * blockDim.x + threadIdx.x;
  if (id >= Bq * Hq) return;
  int b = id / Hq, hh = id % Hq;
  const float* w  = W_attn + (size_t)hh * (2 * Hq) + Hq;   // second-half columns
  const float* hb = h + (size_t)b * Hq;
  float a = 0.f;
  for (int k = 0; k < Hq; ++k) a += w[k] * hb[k];
  q[id] = a;
}

__global__ void k_attn_step(const float* __restrict__ P, const float* __restrict__ q,
                            const float* __restrict__ v_attn, float* __restrict__ attns, int t) {
  int wid = (blockIdx.x * blockDim.x + threadIdx.x) >> 5;   // node id = b*NN + n
  int lane = threadIdx.x & 31;
  if (wid >= Bq * NNq) return;
  int b = wid / NNq, n = wid % NNq;
  const float* pr = P + (size_t)wid * Hq;
  const float* qb = q + (size_t)b * Hq;
  float a = 0.f;
  for (int j = lane; j < Hq; j += 32) a += v_attn[j] * tanhf(pr[j] + qb[j]);
  for (int o = 16; o > 0; o >>= 1) a += __shfl_down(a, o, 32);
  if (lane == 0) attns[((size_t)t * Bq + b) * NNq + n] = a;
}

__global__ void k_pi(const float* __restrict__ attns, const int* __restrict__ vnr_VNF,
                     const int* __restrict__ net_VNF, float* __restrict__ pi) {
  int id = blockIdx.x * blockDim.x + threadIdx.x;
  if (id >= Bq * NVq * NNq) return;
  int b = id / (NVq * NNq);
  int rem = id % (NVq * NNq);
  int t = rem / NNq, n = rem % NNq;
  int vt = vnr_VNF[b * NVq + t];
  int mv = net_VNF[((size_t)b * NNq + n) * Tq + vt];
  pi[id] = attns[((size_t)t * Bq + b) * NNq + n] + logf((float)mv);
}

__global__ void k_value_reduce(const float* __restrict__ hv3, const float* __restrict__ hs3,
                               const float* __restrict__ nmask, float* __restrict__ y) {
  int b = blockIdx.x;
  int k = threadIdx.x;        // 128 threads
  float s1 = 0.f, ns = 0.f;
  for (int t = 0; t < NVq; ++t) {
    float m = nmask[b * NVq + t];
    s1 += hv3[((size_t)b * NVq + t) * Hq + k] * m;
    ns += m;
  }
  float s2 = 0.f;
  for (int n = 0; n < NNq; ++n) s2 += hs3[((size_t)b * NNq + n) * Hq + k];
  y[(size_t)b * 2 * Hq + k] = s1 / ns;
  y[(size_t)b * 2 * Hq + Hq + k] = s2 * (1.f / (float)NNq);
}

__global__ void k_value_out(const float* __restrict__ y, const float* __restrict__ W_out,
                            const float* __restrict__ b_out, float* __restrict__ val) {
  int id = blockIdx.x * blockDim.x + threadIdx.x;
  if (id >= Bq * Oq) return;
  int b = id / Oq, o = id % Oq;
  const float* yb = y + (size_t)b * 2 * Hq;
  const float* w  = W_out + (size_t)o * 2 * Hq;
  float a = b_out[o];
  for (int k = 0; k < 2 * Hq; ++k) a += yb[k] * w[k];
  val[id] = a;
}

// ---------------- host orchestration ----------------
static inline int ceil_div(int a, int b) { return (a + b - 1) / b; }

static void gat_layer(hipStream_t st, const float* h_in, int Fin, int Kp,
                      const float* W, const float* al, const float* ar, const float* bias,
                      const int* src, const int* dst, int M, int E,
                      __bf16* Abf, __bf16* Wbf, float* z,
                      float* el, float* er, float* emax, float* denom, float* ebuf,
                      float* out) {
  k_convert_pad<<<ceil_div(M * Kp, 256), 256, 0, st>>>(h_in, Fin, Abf, Kp, M * Kp);
  k_pack_wt<<<ceil_div(Hq * Kp, 256), 256, 0, st>>>(W, Fin, Hq, Wbf, Kp);
  k_wmma_gemm<<<M / 128, 256, 0, st>>>(Abf, Wbf, z, M, Kp);
  k_node_dots<<<ceil_div(M, 8), 256, 0, st>>>(z, al, ar, el, er, M);
  k_fill<<<ceil_div(M, 256), 256, 0, st>>>(emax, -3.402823466e38f, M);
  k_fill<<<ceil_div(M, 256), 256, 0, st>>>(denom, 0.f, M);
  k_init_bias<<<ceil_div(M * Hq, 256), 256, 0, st>>>(out, bias, M * Hq);
  k_edge_scores<<<ceil_div(E, 256), 256, 0, st>>>(src, dst, el, er, ebuf, emax, E);
  k_edge_exp<<<ceil_div(E, 256), 256, 0, st>>>(dst, ebuf, emax, denom, E);
  k_edge_aggregate<<<ceil_div(E, 8), 256, 0, st>>>(src, dst, ebuf, denom, z, out, E);
}

extern "C" void kernel_launch(void* const* d_in, const int* in_sizes, int n_in,
                              void* d_out, int out_size, void* d_ws, size_t ws_size,
                              hipStream_t stream) {
  // ---- inputs (setup_inputs order) ----
  const float* static_feat = (const float*)d_in[0];
  const float* vnr_feat    = (const float*)d_in[1];
  const float* nmask       = (const float*)d_in[2];
  const float *Ws1=(const float*)d_in[3],  *als1=(const float*)d_in[4],  *ars1=(const float*)d_in[5],  *bs1=(const float*)d_in[6];
  const float *Ws2=(const float*)d_in[7],  *als2=(const float*)d_in[8],  *ars2=(const float*)d_in[9],  *bs2=(const float*)d_in[10];
  const float *Ws3=(const float*)d_in[11], *als3=(const float*)d_in[12], *ars3=(const float*)d_in[13], *bs3=(const float*)d_in[14];
  const float *Wv1=(const float*)d_in[15], *alv1=(const float*)d_in[16], *arv1=(const float*)d_in[17], *bv1=(const float*)d_in[18];
  const float *Wv2=(const float*)d_in[19], *alv2=(const float*)d_in[20], *arv2=(const float*)d_in[21], *bv2=(const float*)d_in[22];
  const float *Wv3=(const float*)d_in[23], *alv3=(const float*)d_in[24], *arv3=(const float*)d_in[25], *bv3=(const float*)d_in[26];
  const float* W_attn = (const float*)d_in[27];
  const float* v_attn = (const float*)d_in[28];
  const float* W_ih   = (const float*)d_in[29];
  const float* W_hh   = (const float*)d_in[30];
  const float* b_ih   = (const float*)d_in[31];
  const float* b_hh   = (const float*)d_in[32];
  const float* h0     = (const float*)d_in[33];
  const float* W_out  = (const float*)d_in[34];
  const float* b_out  = (const float*)d_in[35];
  const int* src_s = (const int*)d_in[36];
  const int* dst_s = (const int*)d_in[37];
  const int* src_v = (const int*)d_in[38];
  const int* dst_v = (const int*)d_in[39];
  const int* vnr_VNF = (const int*)d_in[40];
  const int* net_VNF = (const int*)d_in[41];

  // ---- workspace carve ----
  char* p = (char*)d_ws;
  auto carve = [&](size_t bytes) { void* r = (void*)p; p += (bytes + 255) & ~(size_t)255; return r; };
  const int MS = Bq * NNq;            // 64000
  const int MV = Bq * NVq;            // 640
  float*  Zbuf  = (float*)carve((size_t)MS * Hq * 4);  // per-layer z, later P
  float*  H1    = (float*)carve((size_t)MS * Hq * 4);
  float*  H2    = (float*)carve((size_t)MS * Hq * 4);
  __bf16* Abf   = (__bf16*)carve((size_t)MS * Hq * 2);
  __bf16* Wbf   = (__bf16*)carve((size_t)Hq * Hq * 2);
  float*  el    = (float*)carve((size_t)MS * 4);
  float*  er    = (float*)carve((size_t)MS * 4);
  float*  emax  = (float*)carve((size_t)MS * 4);
  float*  denom = (float*)carve((size_t)MS * 4);
  float*  ebuf  = (float*)carve((size_t)ENq * 4);
  float*  hv1   = (float*)carve((size_t)MV * Hq * 4);
  float*  hv2   = (float*)carve((size_t)MV * Hq * 4);
  float*  attns = (float*)carve((size_t)NVq * Bq * NNq * 4);
  float*  hstate= (float*)carve((size_t)Bq * Hq * 4);
  float*  gi    = (float*)carve((size_t)Bq * 3 * Hq * 4);
  float*  gh    = (float*)carve((size_t)Bq * 3 * Hq * 4);
  float*  q     = (float*)carve((size_t)Bq * Hq * 4);
  float*  ybuf  = (float*)carve((size_t)Bq * 2 * Hq * 4);

  // ---- static graph: 3 GAT layers ----
  gat_layer(stream, static_feat, 8,  32,  Ws1, als1, ars1, bs1, src_s, dst_s, MS, ENq,
            Abf, Wbf, Zbuf, el, er, emax, denom, ebuf, H1);
  gat_layer(stream, H1,          Hq, Hq,  Ws2, als2, ars2, bs2, src_s, dst_s, MS, ENq,
            Abf, Wbf, Zbuf, el, er, emax, denom, ebuf, H2);
  gat_layer(stream, H2,          Hq, Hq,  Ws3, als3, ars3, bs3, src_s, dst_s, MS, ENq,
            Abf, Wbf, Zbuf, el, er, emax, denom, ebuf, H1);
  const float* hs3 = H1;

  // ---- vnr graph: 3 GAT layers (reuse big scratch) ----
  gat_layer(stream, vnr_feat, 6,  32, Wv1, alv1, arv1, bv1, src_v, dst_v, MV, EVq,
            Abf, Wbf, Zbuf, el, er, emax, denom, ebuf, hv1);
  gat_layer(stream, hv1,      Hq, Hq, Wv2, alv2, arv2, bv2, src_v, dst_v, MV, EVq,
            Abf, Wbf, Zbuf, el, er, emax, denom, ebuf, hv2);
  gat_layer(stream, hv2,      Hq, Hq, Wv3, alv3, arv3, bv3, src_v, dst_v, MV, EVq,
            Abf, Wbf, Zbuf, el, er, emax, denom, ebuf, hv1);
  const float* hv3 = hv1;

  // ---- step-invariant attention precompute: P = hs3 @ W_attn[:, :H]^T ----
  k_convert_pad<<<ceil_div(MS * Hq, 256), 256, 0, stream>>>(hs3, Hq, Abf, Hq, MS * Hq);
  k_pack_rows<<<ceil_div(Hq * Hq, 256), 256, 0, stream>>>(W_attn, 2 * Hq, 0, Hq, Hq, Wbf);
  k_wmma_gemm<<<MS / 128, 256, 0, stream>>>(Abf, Wbf, Zbuf, MS, Hq);
  const float* P = Zbuf;

  // ---- GRU pointer decoder ----
  k_h_init<<<ceil_div(Bq * Hq, 256), 256, 0, stream>>>(hstate, h0);
  for (int t = 0; t < NVq; ++t) {
    k_gru_gates<<<ceil_div(Bq * 3 * Hq, 256), 256, 0, stream>>>(hv3, t, hstate, W_ih, W_hh, b_ih, b_hh, gi, gh);
    k_gru_combine<<<ceil_div(Bq * Hq, 256), 256, 0, stream>>>(gi, gh, hstate);
    k_q_proj<<<ceil_div(Bq * Hq, 256), 256, 0, stream>>>(W_attn, hstate, q);
    k_attn_step<<<ceil_div(Bq * NNq, 8), 256, 0, stream>>>(P, q, v_attn, attns, t);
  }

  // ---- outputs ----
  float* pi_out  = (float*)d_out;            // [B, NV*NN] = 640000
  float* val_out = pi_out + Bq * NVq * NNq;  // [B, O]     = 4096
  k_pi<<<ceil_div(Bq * NVq * NNq, 256), 256, 0, stream>>>(attns, vnr_VNF, net_VNF, pi_out);
  k_value_reduce<<<Bq, Hq, 0, stream>>>(hv3, hs3, nmask, ybuf);
  k_value_out<<<ceil_div(Bq * Oq, 256), 256, 0, stream>>>(ybuf, W_out, b_out, val_out);
}
